// Rel_Transformer_Layer_7344394076550
// MI455X (gfx1250) — matrix operand order repaired and checked by reading, not compiled, not run on hardware
//
#include <hip/hip_runtime.h>
#include <hip/hip_bf16.h>

// ---------------------------------------------------------------------------
// MI455X (gfx1250) relational-transformer layer.
//   * all matmuls: v_wmma_f32_16x16x32_bf16, fp32 accumulate
//   * B / K / V tiles staged into LDS with the Tensor Data Mover
//     (tensor_load_to_lds + s_wait_tensorcnt), double-buffered so the DMA
//     overlaps WMMA compute; 8 waves per block share each staged tile.
//   * flash-style online-softmax attention, one wave per 16-row tile/head.
// ---------------------------------------------------------------------------

#define Nn 2048
#define Dd 1024
#define Hh 8
#define DKk 128
#define FFf 4096
#define ATTN_SCALE 0.08838834764831845f  // 1/sqrt(128)

typedef __attribute__((ext_vector_type(16))) __bf16 v16bf;
typedef __attribute__((ext_vector_type(8)))  float  v8f;
typedef __attribute__((ext_vector_type(4)))  unsigned int v4u;
typedef __attribute__((ext_vector_type(8)))  int v8i_;
typedef __attribute__((ext_vector_type(4)))  int v4i_;

union Frag { v16bf v; unsigned int u[8]; };

#if defined(__gfx1250__) && __has_builtin(__builtin_amdgcn_tensor_load_to_lds)
#define HAVE_TDM 1
#else
#define HAVE_TDM 0
#endif

#if HAVE_TDM && __has_builtin(__builtin_amdgcn_s_wait_tensorcnt)
#define TDM_WAIT() __builtin_amdgcn_s_wait_tensorcnt(0)
#else
#define TDM_WAIT()
#endif

// A-matrix (16x32 bf16) K-pattern: vgpr j holds K pair {k,k+1},
// k = 2j + 16*half   (half = lane>>4)
__device__ __forceinline__ int kpatA(int j, int half) {
  return (j << 1) + (half << 4);
}

// ---------------------------------------------------------------------------
// Stage a 2D bf16 tile [rows x cols] (row stride `stride` elements, cols
// contiguous) from global into LDS via the Tensor Data Mover. Call from one
// wave only; completion = TDM_WAIT() by that wave + block barrier.
__device__ __forceinline__ void stage_tile(__bf16* lds, const __bf16* gptr,
                                           int cols, int rows, int stride,
                                           int tdim0, int tdim1, int lane) {
#if HAVE_TDM
  unsigned long long ga = (unsigned long long)(size_t)gptr;
  unsigned la = (unsigned)(size_t)lds;  // LDS aperture: addr[31:0] = LDS byte addr
  v4u g0;
  g0[0] = 1u;                                             // count=1, user mode
  g0[1] = la;                                             // lds_addr
  g0[2] = (unsigned)ga;                                   // global_addr lo
  g0[3] = (unsigned)((ga >> 32) & 0x1FFFFFFu) | (2u << 30);  // addr hi | type=2
  v8i_ g1;
  g1[0] = 0x10000;                                        // data_size=1 (2B)
  g1[1] = (int)((unsigned)(tdim0 & 0xFFFF) << 16);        // tensor_dim0 lo
  g1[2] = (int)(((unsigned)tdim0 >> 16) |
                ((unsigned)(tdim1 & 0xFFFF) << 16));      // td0 hi | td1 lo
  g1[3] = (int)(((unsigned)tdim1 >> 16) |
                ((unsigned)cols << 16));                  // td1 hi | tile_dim0
  g1[4] = rows;                                           // tile_dim1 (dim2=0)
  g1[5] = stride;                                         // dim0_stride lo
  g1[6] = 0;
  g1[7] = 0;
  // 6-arg form (clang-23 / therock-10.0): groups 0..3 + extra group + cpol
  __builtin_amdgcn_tensor_load_to_lds(g0, g1, (v4i_)0, (v4i_)0, (v8i_)0, 0);
  (void)lane;
#else
  for (int i = lane; i < rows * cols; i += 32)
    lds[i] = gptr[(i / cols) * stride + (i % cols)];
#endif
}

// ---------------------------------------------------------------------------
__global__ void etw_kernel(const float* __restrict__ re,
                           const float* __restrict__ rw,
                           float* __restrict__ etw) {
  int t = threadIdx.x;  // 64 threads
  float s = 0.f;
  #pragma unroll 8
  for (int d = 0; d < DKk; ++d) s += re[t * DKk + d] * rw[t * DKk + d];
  etw[t] = s * ATTN_SCALE;
}

__global__ __launch_bounds__(256) void xbias_kernel(
    const float* __restrict__ x, const int* __restrict__ ind,
    const int* __restrict__ outd, const float* __restrict__ ie,
    const float* __restrict__ oe, float* __restrict__ xf,
    __bf16* __restrict__ xb) {
  int idx = blockIdx.x * 256 + threadIdx.x;
  int i = idx >> 10, j = idx & 1023;
  float v = x[idx] + ie[ind[i] * Dd + j] + oe[outd[i] * Dd + j];
  xf[idx] = v;
  xb[idx] = (__bf16)v;
}

__global__ __launch_bounds__(256) void bias2d_kernel(
    const int* __restrict__ dist, const int* __restrict__ pet,
    const float* __restrict__ pw, const float* __restrict__ etw,
    const float* __restrict__ plen, float* __restrict__ out) {
  __shared__ float e[64], pl[16];
  int tid = threadIdx.x;
  if (tid < 64) e[tid] = etw[tid];
  if (tid < 16) pl[tid] = plen[tid];
  __syncthreads();
  int idx = blockIdx.x * 256 + tid;
  float s = pl[dist[idx]];
  #pragma unroll
  for (int l = 0; l < 4; ++l) s += e[pet[idx * 4 + l]] * pw[idx * 4 + l];
  out[idx] = s;
}

__global__ __launch_bounds__(256) void wtrans_kernel(
    const float* __restrict__ W, __bf16* __restrict__ WT, int K, int Nc) {
  int idx = blockIdx.x * 256 + threadIdx.x;
  int k = idx / Nc, n = idx % Nc;
  WT[n * K + k] = (__bf16)W[idx];
}

__global__ __launch_bounds__(256) void ktrans_kernel(
    const __bf16* __restrict__ Kb, __bf16* __restrict__ Kt) {
  int idx = blockIdx.x * 256 + threadIdx.x;       // over N*D
  int h = idx >> 18, r = idx & 262143;
  int n = r >> 7, dk = r & 127;
  Kt[idx] = Kb[((h << 8) + (dk << 1) + (n >> 10)) * 1024 + (n & 1023)];
}

__global__ __launch_bounds__(256) void vtrans_kernel(
    const __bf16* __restrict__ Vb, __bf16* __restrict__ Vt) {
  int idx = blockIdx.x * 256 + threadIdx.x;       // over N*D
  int h = idx >> 18, r = idx & 262143;
  int n = r & 2047, dk = r >> 11;
  Vt[idx] = Vb[(h << 18) + n * 128 + dk];
}

// ---------------------------------------------------------------------------
// bf16 WMMA GEMM: C[M,Nc] = A[M,K] @ B[K,Nc] (+bias)(+resid)(ReLU?)
// BT = B^T [Nc,K] bf16. Wave tile 16x64; all 8 waves of a block share one
// 64x32 B tile, TDM-staged into LDS and double-buffered. A fragments are
// software-pipelined global b128 loads.
__global__ __launch_bounds__(256) void gemm_bf16_kernel(
    const __bf16* __restrict__ A, const __bf16* __restrict__ BT,
    const float* __restrict__ bias, const float* __restrict__ resid,
    float* __restrict__ outf, __bf16* __restrict__ outb,
    int M, int Nc, int K, int relu) {
  __shared__ __bf16 Bl[2][64][32];

  int tid = threadIdx.x;
  int wslot = tid >> 5;
  int lane = tid & 31;
  int Mt = M >> 4;                          // multiple of 8 for all our shapes
  int wid = blockIdx.x * 8 + wslot;
  int mt = wid % Mt, nt = wid / Mt;         // nt uniform across the block
  int m0 = mt << 4, n0 = nt << 6;
  int nl = lane & 15, half = lane >> 4;

  v8f acc[4] = {};
  const unsigned int* arow = (const unsigned int*)(A + (m0 + nl) * K);

  // prologue: stage B tile for k0=0
  if (wslot == 0) {
    stage_tile(&Bl[0][0][0], BT + n0 * K, 32, 64, K, K, Nc, lane);
    TDM_WAIT();
  }
  Frag a_cur;
  #pragma unroll
  for (int j = 0; j < 8; ++j) a_cur.u[j] = arow[kpatA(j, half) >> 1];
  __syncthreads();

  for (int k0 = 0; k0 < K; k0 += 32) {
    int buf = (k0 >> 5) & 1;
    bool more = (k0 + 32) < K;
    if (wslot == 0 && more)
      stage_tile(&Bl[buf ^ 1][0][0], BT + n0 * K + k0 + 32, 32, 64, K, K, Nc,
                 lane);
    Frag a_nxt;
    if (more) {
      #pragma unroll
      for (int j = 0; j < 8; ++j)
        a_nxt.u[j] = arow[(k0 + 32 + kpatA(j, half)) >> 1];
      __builtin_prefetch((const void*)(arow + ((k0 + 64) >> 1)), 0, 0);
    }
    #pragma unroll
    for (int t = 0; t < 4; ++t) {
      Frag b;
      const unsigned int* bp =
          (const unsigned int*)(&Bl[buf][(t << 4) + nl][half << 4]);
      #pragma unroll
      for (int j = 0; j < 8; ++j) b.u[j] = bp[j];
      acc[t] = __builtin_amdgcn_wmma_f32_16x16x32_bf16(
          false, a_cur.v, false, b.v, (short)0, acc[t], false, false);
    }
    if (more) {
      if (wslot == 0) TDM_WAIT();
      a_cur = a_nxt;
      __syncthreads();
    }
  }
  #pragma unroll
  for (int t = 0; t < 4; ++t) {
    int n = n0 + (t << 4) + nl;
    float bv = bias ? bias[n] : 0.f;
    #pragma unroll
    for (int vv = 0; vv < 8; ++vv) {
      int m = m0 + vv + (half << 3);
      float val = acc[t][vv] + bv;
      if (resid) val += resid[m * Nc + n];
      if (relu)  val = fmaxf(val, 0.f);
      if (outf)  outf[m * Nc + n] = val;
      if (outb)  outb[m * Nc + n] = (__bf16)val;
    }
  }
}

// ---------------------------------------------------------------------------
// Flash attention. One wave per (head, 16-row tile); the 8 waves of a block
// cover consecutive row tiles of the SAME head, so they share each 32-column
// K/V tile: those are TDM-staged into LDS, double-buffered.
__global__ __launch_bounds__(256) void attn_kernel(
    const __bf16* __restrict__ Qbf, const __bf16* __restrict__ Kt,
    const __bf16* __restrict__ Vt, const float* __restrict__ bias2d,
    __bf16* __restrict__ Obf) {
  __shared__ __bf16 Kl[2][32][128];       // [token][dk]
  __shared__ __bf16 Vl[2][128][32];       // [dk][token]
  __shared__ __bf16 Pl[8][16 * 40];       // per-wave P tile, padded rows
  __shared__ float  Ml[8][16], Al[8][16], Ll[8][16];

  int wslot = threadIdx.x >> 5;
  int lane  = threadIdx.x & 31;
  int gid   = blockIdx.x * 8 + wslot;
  int h  = gid >> 7;                      // 128 row-tiles per head
  int rt = gid & 127;
  int row0 = rt << 4;
  int nl = lane & 15, half = lane >> 4;

  const __bf16* Qh = Qbf + (h << 18);     // [2048,128] row-major
  const __bf16* Kh = Kt  + (h << 18);     // [n][dk]
  const __bf16* Vh = Vt  + (h << 18);     // [dk][n]
  __bf16*       Oh = Obf + (h << 18);

  // Q fragments held for the whole pass: 4 x (16x32)
  Frag qf[4];
  const unsigned int* qp = (const unsigned int*)(Qh + (row0 + nl) * 128);
  #pragma unroll
  for (int d = 0; d < 4; ++d)
    #pragma unroll
    for (int j = 0; j < 8; ++j)
      qf[d].u[j] = qp[(d * 32 + kpatA(j, half)) >> 1];

  if (lane < 16) { Ml[wslot][lane] = -1e30f; Ll[wslot][lane] = 0.f; }
  v8f acc[8] = {};                        // O^T: 8 frags over dk

  // prologue: stage K/V tiles for c0=0
  if (wslot == 0) {
    stage_tile(&Kl[0][0][0], Kh, 128, 32, 128, 128, Nn, lane);
    stage_tile(&Vl[0][0][0], Vh, 32, 128, Nn, Nn, DKk, lane);
    TDM_WAIT();
  }
  __syncthreads();

  for (int c0 = 0; c0 < Nn; c0 += 32) {
    int buf = (c0 >> 5) & 1;
    bool more = (c0 + 32) < Nn;
    if (wslot == 0 && more) {
      stage_tile(&Kl[buf ^ 1][0][0], Kh + (c0 + 32) * 128, 128, 32, 128, 128,
                 Nn, lane);
      stage_tile(&Vl[buf ^ 1][0][0], Vh + (c0 + 32), 32, 128, Nn, Nn, DKk,
                 lane);
    }
    // ---- scores for two 16-col subtiles, scale + bias fused
    v8f s[2];
    #pragma unroll
    for (int sub = 0; sub < 2; ++sub) {
      v8f sv = {};
      #pragma unroll
      for (int d = 0; d < 4; ++d) {
        Frag b;
        const unsigned int* kp = (const unsigned int*)(
            &Kl[buf][sub * 16 + nl][d * 32 + (half << 4)]);
        #pragma unroll
        for (int j = 0; j < 8; ++j) b.u[j] = kp[j];
        sv = __builtin_amdgcn_wmma_f32_16x16x32_bf16(
            false, qf[d].v, false, b.v, (short)0, sv, false, false);
      }
      const float* bp = bias2d + (row0 + (half << 3)) * Nn + c0 + sub * 16 + nl;
      #pragma unroll
      for (int vv = 0; vv < 8; ++vv)
        sv[vv] = sv[vv] * ATTN_SCALE + bp[vv * Nn];
      s[sub] = sv;
    }
    // ---- per-row tile max (cross-lane reduce within 16-lane groups)
    float tmax[8];
    #pragma unroll
    for (int vv = 0; vv < 8; ++vv) tmax[vv] = fmaxf(s[0][vv], s[1][vv]);
    #pragma unroll
    for (int m = 1; m < 16; m <<= 1)
      #pragma unroll
      for (int vv = 0; vv < 8; ++vv)
        tmax[vv] = fmaxf(tmax[vv], __shfl_xor(tmax[vv], m, 32));
    __syncthreads();
    if (nl == 0) {  // lanes 0 and 16 own rows [0..7] / [8..15]
      #pragma unroll
      for (int vv = 0; vv < 8; ++vv) {
        int r = vv + (half << 3);
        float mo = Ml[wslot][r];
        float mn = fmaxf(mo, tmax[vv]);
        float al = __expf(mo - mn);
        Ml[wslot][r] = mn; Al[wslot][r] = al;
        Ll[wslot][r] *= al;
      }
    }
    __syncthreads();
    // ---- exp, row-sum, write P (bf16) to LDS
    float rsum[8];
    #pragma unroll
    for (int vv = 0; vv < 8; ++vv) {
      int r = vv + (half << 3);
      float mn = Ml[wslot][r];
      float p0 = __expf(s[0][vv] - mn);
      float p1 = __expf(s[1][vv] - mn);
      rsum[vv] = p0 + p1;
      Pl[wslot][r * 40 + nl]      = (__bf16)p0;
      Pl[wslot][r * 40 + 16 + nl] = (__bf16)p1;
    }
    #pragma unroll
    for (int m = 1; m < 16; m <<= 1)
      #pragma unroll
      for (int vv = 0; vv < 8; ++vv)
        rsum[vv] += __shfl_xor(rsum[vv], m, 32);
    if (nl == 0) {
      #pragma unroll
      for (int vv = 0; vv < 8; ++vv) Ll[wslot][vv + (half << 3)] += rsum[vv];
    }
    __syncthreads();
    // ---- rescale O^T accumulators: column N = lane%16 = output row
    float al = Al[wslot][nl];
    #pragma unroll
    for (int t = 0; t < 8; ++t)
      #pragma unroll
      for (int vv = 0; vv < 8; ++vv) acc[t][vv] *= al;
    // ---- P^T as B fragment from LDS
    Frag pb;
    const unsigned int* pp =
        (const unsigned int*)(&Pl[wslot][nl * 40 + (half << 4)]);
    #pragma unroll
    for (int j = 0; j < 8; ++j) pb.u[j] = pp[j];
    // ---- acc += V^T_tile @ P^T   (8 WMMAs over dk)
    #pragma unroll
    for (int t = 0; t < 8; ++t) {
      Frag va;
      const unsigned int* vp =
          (const unsigned int*)(&Vl[buf][t * 16 + nl][0]);
      #pragma unroll
      for (int j = 0; j < 8; ++j) va.u[j] = vp[kpatA(j, half) >> 1];
      acc[t] = __builtin_amdgcn_wmma_f32_16x16x32_bf16(
          false, va.v, false, pb.v, (short)0, acc[t], false, false);
    }
    if (wslot == 0 && more) TDM_WAIT();
    __syncthreads();
  }
  // ---- epilogue: divide by row sums, store O (bf16, natural layout)
  float inv = 1.0f / Ll[wslot][nl];
  #pragma unroll
  for (int t = 0; t < 8; ++t)
    #pragma unroll
    for (int vv = 0; vv < 8; ++vv) {
      int dk = t * 16 + vv + (half << 3);
      Oh[(row0 + nl) * 128 + dk] = (__bf16)(acc[t][vv] * inv);
    }
}

// ---------------------------------------------------------------------------
__global__ __launch_bounds__(256) void ln_kernel(
    const float* __restrict__ in, const float* __restrict__ g,
    const float* __restrict__ bt, float* __restrict__ outf,
    __bf16* __restrict__ outb) {
  __shared__ float red[256];
  int row = blockIdx.x, tid = threadIdx.x;
  const float* r = in + row * Dd;
  float lv[4], s = 0.f;
  #pragma unroll
  for (int i = 0; i < 4; ++i) { lv[i] = r[tid + 256 * i]; s += lv[i]; }
  red[tid] = s; __syncthreads();
  for (int o = 128; o > 0; o >>= 1) {
    if (tid < o) red[tid] += red[tid + o];
    __syncthreads();
  }
  float mean = red[0] * (1.0f / Dd);
  __syncthreads();
  float vs = 0.f;
  #pragma unroll
  for (int i = 0; i < 4; ++i) { float d = lv[i] - mean; vs += d * d; }
  red[tid] = vs; __syncthreads();
  for (int o = 128; o > 0; o >>= 1) {
    if (tid < o) red[tid] += red[tid + o];
    __syncthreads();
  }
  float inv = rsqrtf(red[0] * (1.0f / Dd) + 1e-5f);
  #pragma unroll
  for (int i = 0; i < 4; ++i) {
    int c = tid + 256 * i;
    float o = (lv[i] - mean) * inv * g[c] + bt[c];
    outf[row * Dd + c] = o;
    if (outb) outb[row * Dd + c] = (__bf16)o;
  }
}

// ---------------------------------------------------------------------------
extern "C" void kernel_launch(void* const* d_in, const int* in_sizes, int n_in,
                              void* d_out, int out_size, void* d_ws,
                              size_t ws_size, hipStream_t stream) {
  const float* x    = (const float*)d_in[0];
  const float* lgx  = (const float*)d_in[1];
  const int*   ind  = (const int*)d_in[2];
  const int*   outd = (const int*)d_in[3];
  const int*   dist = (const int*)d_in[4];
  const int*   pet  = (const int*)d_in[5];
  const float* pw   = (const float*)d_in[6];
  const float* rew  = (const float*)d_in[7];
  const float* rww  = (const float*)d_in[8];
  const float* ide  = (const float*)d_in[9];
  const float* ode  = (const float*)d_in[10];
  const float* plen = (const float*)d_in[11];
  const float* Wq = (const float*)d_in[12]; const float* bq = (const float*)d_in[13];
  const float* Wk = (const float*)d_in[14]; const float* bk = (const float*)d_in[15];
  const float* Wv = (const float*)d_in[16]; const float* bv = (const float*)d_in[17];
  const float* Wo = (const float*)d_in[18]; const float* bo = (const float*)d_in[19];
  const float* l1g = (const float*)d_in[20]; const float* l1b = (const float*)d_in[21];
  const float* W1 = (const float*)d_in[22]; const float* b1 = (const float*)d_in[23];
  const float* W2 = (const float*)d_in[24]; const float* b2 = (const float*)d_in[25];
  const float* l2g = (const float*)d_in[26]; const float* l2b = (const float*)d_in[27];

  size_t off = 0;
  auto alloc = [&](size_t bytes) -> void* {
    off = (off + 255) & ~(size_t)255;
    void* p = (char*)d_ws + off;
    off += bytes;
    return p;
  };
  const size_t ND = (size_t)Nn * Dd;
  float*  etw  = (float*)alloc(64 * 4);
  float*  xf   = (float*)alloc(ND * 4);
  __bf16* xb   = (__bf16*)alloc(ND * 2);
  __bf16* WqT  = (__bf16*)alloc((size_t)Dd * Dd * 2);
  __bf16* WkT  = (__bf16*)alloc((size_t)Dd * Dd * 2);
  __bf16* WvT  = (__bf16*)alloc((size_t)Dd * Dd * 2);
  __bf16* WoT  = (__bf16*)alloc((size_t)Dd * Dd * 2);
  __bf16* W1T  = (__bf16*)alloc((size_t)Dd * FFf * 2);
  __bf16* W2T  = (__bf16*)alloc((size_t)Dd * FFf * 2);
  __bf16* Qb   = (__bf16*)alloc(ND * 2);
  __bf16* Kb   = (__bf16*)alloc(ND * 2);
  __bf16* Vb   = (__bf16*)alloc(ND * 2);
  __bf16* Ktr  = (__bf16*)alloc(ND * 2);
  __bf16* Vtr  = (__bf16*)alloc(ND * 2);
  float*  b2d  = (float*)alloc((size_t)Nn * Nn * 4);
  __bf16* Ob   = (__bf16*)alloc(ND * 2);
  float*  pre1 = (float*)alloc(ND * 4);
  float*  hf   = (float*)alloc(ND * 4);
  __bf16* hb   = (__bf16*)alloc(ND * 2);
  __bf16* Fb   = (__bf16*)alloc((size_t)Nn * FFf * 2);
  float*  pre2 = (float*)alloc(ND * 4);

  // structural biases
  etw_kernel<<<1, 64, 0, stream>>>(rew, rww, etw);
  xbias_kernel<<<ND / 256, 256, 0, stream>>>(x, ind, outd, ide, ode, xf, xb);
  bias2d_kernel<<<(size_t)Nn * Nn / 256, 256, 0, stream>>>(dist, pet, pw, etw, plen, b2d);

  // weight convert+transpose to bf16
  wtrans_kernel<<<Dd * Dd / 256, 256, 0, stream>>>(Wq, WqT, Dd, Dd);
  wtrans_kernel<<<Dd * Dd / 256, 256, 0, stream>>>(Wk, WkT, Dd, Dd);
  wtrans_kernel<<<Dd * Dd / 256, 256, 0, stream>>>(Wv, WvT, Dd, Dd);
  wtrans_kernel<<<Dd * Dd / 256, 256, 0, stream>>>(Wo, WoT, Dd, Dd);
  wtrans_kernel<<<Dd * FFf / 256, 256, 0, stream>>>(W1, W1T, Dd, FFf);
  wtrans_kernel<<<Dd * FFf / 256, 256, 0, stream>>>(W2, W2T, FFf, Dd);

  // Q/K/V projections (bf16 out)
  int blkP = (Nn / 16) * (Dd / 64) / 8;  // 256 blocks
  gemm_bf16_kernel<<<blkP, 256, 0, stream>>>(xb, WqT, bq, nullptr, nullptr, Qb, Nn, Dd, Dd, 0);
  gemm_bf16_kernel<<<blkP, 256, 0, stream>>>(xb, WkT, bk, nullptr, nullptr, Kb, Nn, Dd, Dd, 0);
  gemm_bf16_kernel<<<blkP, 256, 0, stream>>>(xb, WvT, bv, nullptr, nullptr, Vb, Nn, Dd, Dd, 0);
  ktrans_kernel<<<ND / 256, 256, 0, stream>>>(Kb, Ktr);
  vtrans_kernel<<<ND / 256, 256, 0, stream>>>(Vb, Vtr);

  // attention (1024 waves = 128 blocks)
  attn_kernel<<<128, 256, 0, stream>>>(Qb, Ktr, Vtr, b2d, Ob);

  // output projection + residual, LN1
  gemm_bf16_kernel<<<blkP, 256, 0, stream>>>(Ob, WoT, bo, xf, pre1, nullptr, Nn, Dd, Dd, 0);
  ln_kernel<<<Nn, 256, 0, stream>>>(pre1, l1g, l1b, hf, hb);

  // FFN
  int blkF1 = (Nn / 16) * (FFf / 64) / 8;  // 1024 blocks
  gemm_bf16_kernel<<<blkF1, 256, 0, stream>>>(hb, W1T, b1, nullptr, nullptr, Fb, Nn, FFf, Dd, 1);
  gemm_bf16_kernel<<<blkP, 256, 0, stream>>>(Fb, W2T, b2, hf, pre2, nullptr, Nn, Dd, FFf, 0);
  ln_kernel<<<Nn, 256, 0, stream>>>(pre2, l2g, l2b, (float*)d_out, nullptr);

  // lgx passthrough (second tuple element)
  (void)hipMemcpyAsync((float*)d_out + ND, lgx,
                       (size_t)Nn * 8 * Dd * sizeof(float),
                       hipMemcpyDeviceToDevice, stream);
}
